// DsFormer_69870527971756
// MI455X (gfx1250) — matrix operand order repaired and checked by reading, not compiled, not run on hardware
//
#include <hip/hip_runtime.h>
#include <hip/hip_bf16.h>
#include <math.h>

// ---------------------------------------------------------------------------
// DsFormer forward for MI455X (gfx1250, wave32, WMMA).
// GEMMs: v_wmma_f32_16x16x32_f16 with DOUBLE-BUFFERED async global->LDS
// staging (GLOBAL_LOAD_ASYNC_TO_LDS_B32 + s_wait_asynccnt), the CDNA5
// cp.async-style pipeline. ProbSparse attention implemented as its dense
// u=L equivalent (JAX RNG bits are not reproducible on-device).
//
// Workspace requirement: ~153 MB (see layout in kernel_launch).
// ---------------------------------------------------------------------------

typedef __attribute__((ext_vector_type(16))) _Float16 v16h;
typedef __attribute__((ext_vector_type(8)))  float    v8f;

#define BB   16
#define TT   512
#define FF   21
#define DD   512
#define HH   8
#define DH   64
#define DFFN 2048
#define NLAY 3
#define MAW  25
#define MPAD 12

// ======================= 1. series decomposition + trend ====================
__global__ void decomp_trend_kernel(const float* __restrict__ x,
                                    const float* __restrict__ trend_w,
                                    const float* __restrict__ trend_b,
                                    float* __restrict__ seasonal,
                                    float* __restrict__ trend_out) {
  int b = blockIdx.x;
  int f = threadIdx.x;
  if (f >= FF) return;
  // moving average window of 25 over edge-replicated series
  float win = 0.f;
  for (int i = -MPAD; i <= MPAD; ++i) {
    int ti = min(max(i, 0), TT - 1);
    win += x[(b * TT + ti) * FF + f];
  }
  float tacc = 0.f;
  for (int t = 0; t < TT; ++t) {
    float mm = win * (1.0f / MAW);
    seasonal[(b * TT + t) * FF + f] = x[(b * TT + t) * FF + f] - mm;
    tacc += mm * trend_w[t];                  // trend_w shape (1, T)
    int lo = max(t - MPAD, 0);
    int hi = min(t + MPAD + 1, TT - 1);
    win += x[(b * TT + hi) * FF + f] - x[(b * TT + lo) * FF + f];
  }
  trend_out[b * FF + f] = tacc + trend_b[0];
}

// ================= 2. circular-pad conv1d (F->D, k=3) + pos-emb =============
__global__ __launch_bounds__(256) void conv_pos_kernel(
    const float* __restrict__ seasonal, const float* __restrict__ conv_w,
    float* __restrict__ enc) {
  int bt = blockIdx.x;                        // 0..B*T-1
  int b = bt >> 9, t = bt & (TT - 1);
  __shared__ float sx[3][FF];
  int tid = threadIdx.x;
  if (tid < 3 * FF) {
    int j = tid / FF, f = tid % FF;
    int ti = (t + j - 1 + TT) & (TT - 1);     // xp wraps: front=last row, back=first row
    sx[j][f] = seasonal[(b * TT + ti) * FF + f];
  }
  __syncthreads();
  for (int d = tid; d < DD; d += 256) {
    float s = 0.f;
    #pragma unroll
    for (int j = 0; j < 3; ++j)
      for (int f = 0; f < FF; ++f)
        s += sx[j][f] * conv_w[(d * FF + f) * 3 + j];   // OIW layout
    // sinusoidal positional embedding
    int i2 = d >> 1;
    float div = __expf((float)(2 * i2) * (-9.210340371976184f / (float)DD));
    float ang = (float)t * div;
    float pe = (d & 1) ? __cosf(ang) : __sinf(ang);
    enc[bt * DD + d] = s + pe;
  }
}

// ======================= 3. WMMA GEMM: C = act(A @ W^T + bias) ==============
// A: M x K (row-major, lda), W: N x K (row-major, ldw), C: M x N (ldc).
// Block tile 64x128, 256 threads = 8 waves; wave w -> rows (w&3)*16,
// cols (w>>2)*64 (4 WMMA n-subtiles, A-fragment reused 4x).
//
// Staging: double-buffered async global->LDS (ASYNCcnt). Each wave issues
// 8 (A) + 16 (B) = 24 GLOBAL_LOAD_ASYNC_TO_LDS_B32 per K-tile; since async
// ops retire in order, "s_wait_asynccnt <= 24" after issuing tile k+1
// guarantees tile k has fully landed in LDS.
#define BM 64
#define BN 128
#define BK 32
#define BKP 36   // padded K stride (floats) in LDS

template <int ACT>   // 0 = none, 1 = exact GELU, 2 = ReLU
__global__ __launch_bounds__(256) void gemm_bias_act_kernel(
    const float* __restrict__ A, int lda,
    const float* __restrict__ W, int ldw,
    const float* __restrict__ bias,
    float* __restrict__ C, int ldc, int K) {
  __shared__ float sA[2][BM][BKP];   // 2 * 64*36*4  = 18 KB
  __shared__ float sB[2][BN][BKP];   // 2 * 128*36*4 = 36 KB

  const int tid = threadIdx.x;
  const int wave = tid >> 5;
  const int lane = tid & 31;
  const int m0 = blockIdx.x * BM;
  const int n0 = blockIdx.y * BN;
  const int wm = (wave & 3) * 16;
  const int wn = (wave >> 2) * 64;
  const int lrow = lane & 15;
  const int khalf = (lane >> 4) * 8;   // unused; A also reads the contiguous kb range
  const int kb = (lane >> 4) * 16;     // B: contiguous K range per lane group
  const int rbase = (lane >> 4) * 8;   // C: M-half select

  // issue one K-tile's async copies into LDS buffer `buf`
  auto issue_tile = [&](int k0, int buf) {
    #pragma unroll
    for (int it = 0; it < 8; ++it) {               // A: 64x32 = 2048 elems
      const int i = tid + it * 256;
      const int r = i >> 5, c = i & (BK - 1);
      const float* gp = &A[(size_t)(m0 + r) * lda + k0 + c];
      const unsigned loff = (unsigned)(uintptr_t)&sA[buf][r][c];
      asm volatile("global_load_async_to_lds_b32 %0, %1, off"
                   :: "v"(loff), "v"(gp) : "memory");
    }
    #pragma unroll
    for (int it = 0; it < 16; ++it) {              // B: 128x32 = 4096 elems
      const int i = tid + it * 256;
      const int r = i >> 5, c = i & (BK - 1);
      const float* gp = &W[(size_t)(n0 + r) * ldw + k0 + c];
      const unsigned loff = (unsigned)(uintptr_t)&sB[buf][r][c];
      asm volatile("global_load_async_to_lds_b32 %0, %1, off"
                   :: "v"(loff), "v"(gp) : "memory");
    }
  };

  v8f acc[4] = {};
  const int nk = K / BK;
  issue_tile(0, 0);

  for (int kt = 0; kt < nk; ++kt) {
    const int buf = kt & 1;
    if (kt + 1 < nk) {
      issue_tile((kt + 1) * BK, buf ^ 1);          // prefetch next tile
      asm volatile("s_wait_asynccnt 0x18" ::: "memory");   // <=24 outstanding
    } else {
      asm volatile("s_wait_asynccnt 0x0" ::: "memory");
    }
    __syncthreads();                               // all waves' copies visible

    v16h afrag;
    #pragma unroll
    for (int e = 0; e < 8; ++e) {
      afrag[e]     = (_Float16)sA[buf][wm + lrow][kb + e];
      afrag[e + 8] = (_Float16)sA[buf][wm + lrow][kb + 8 + e];
    }
    #pragma unroll
    for (int nt = 0; nt < 4; ++nt) {
      v16h bfrag;
      const int col = wn + nt * 16 + lrow;
      #pragma unroll
      for (int e = 0; e < 16; ++e)
        bfrag[e] = (_Float16)sB[buf][col][kb + e];
      acc[nt] = __builtin_amdgcn_wmma_f32_16x16x32_f16(
          false, afrag, false, bfrag, (short)0, acc[nt], false, false);
    }
    __syncthreads();                               // done reading buf
  }

  #pragma unroll
  for (int nt = 0; nt < 4; ++nt) {
    const int col = n0 + wn + nt * 16 + lrow;
    const float bv = bias[col];
    #pragma unroll
    for (int r = 0; r < 8; ++r) {
      const int row = m0 + wm + rbase + r;
      float v = acc[nt][r] + bv;
      if (ACT == 1) v = 0.5f * v * (1.0f + erff(v * 0.70710678118654752f));
      else if (ACT == 2) v = fmaxf(v, 0.0f);
      C[(size_t)row * ldc + col] = v;
    }
  }
}

// ======================= 4. attention (dense softmax, WMMA) =================
// grid (T/16, H, B), one wave per block. Q/K/V/ctx layout: (B*T, D) with the
// head slice at column h*64.
__global__ __launch_bounds__(32) void attention_kernel(
    const float* __restrict__ Q, const float* __restrict__ Km,
    const float* __restrict__ V, float* __restrict__ ctx) {
  __shared__ float    sS[16][TT + 4];
  __shared__ _Float16 sP[16][TT + 8];

  const int qt = blockIdx.x, h = blockIdx.y, b = blockIdx.z;
  const int lane = threadIdx.x;
  const int lrow = lane & 15;
  const int khalf = (lane >> 4) * 8;
  const int kb = (lane >> 4) * 16;
  const int rbase = (lane >> 4) * 8;
  const int base = b * TT * DD + h * DH;      // (b, t=0, h*64)

  // Q fragments (16 rows x 64 K), reused across all 32 key tiles
  v16h qa[2];
  const int qrow = qt * 16 + lrow;
  #pragma unroll
  for (int ks = 0; ks < 2; ++ks) {
    const int k0 = ks * 32;
    #pragma unroll
    for (int e = 0; e < 8; ++e) {
      qa[ks][e]     = (_Float16)Q[base + qrow * DD + k0 + kb + e];
      qa[ks][e + 8] = (_Float16)Q[base + qrow * DD + k0 + kb + 8 + e];
    }
  }

  // scores = Q K^T / sqrt(dh)
  for (int nt = 0; nt < TT / 16; ++nt) {
    v8f acc = {};
    const int ncol = nt * 16 + lrow;
    #pragma unroll
    for (int ks = 0; ks < 2; ++ks) {
      const int k0 = ks * 32;
      v16h bf;
      #pragma unroll
      for (int e = 0; e < 16; ++e)
        bf[e] = (_Float16)Km[base + ncol * DD + k0 + kb + e];
      acc = __builtin_amdgcn_wmma_f32_16x16x32_f16(
          false, qa[ks], false, bf, (short)0, acc, false, false);
    }
    #pragma unroll
    for (int r = 0; r < 8; ++r)
      sS[rbase + r][nt * 16 + lrow] = acc[r] * 0.125f;   // 1/sqrt(64)
  }
  __syncthreads();

  // row softmax, normalized probabilities stored as f16
  if (lane < 16) {
    float mx = -1e30f;
    for (int c = 0; c < TT; ++c) mx = fmaxf(mx, sS[lane][c]);
    float sum = 0.f;
    for (int c = 0; c < TT; ++c) {
      float e = __expf(sS[lane][c] - mx);
      sS[lane][c] = e;
      sum += e;
    }
    const float inv = 1.0f / sum;
    for (int c = 0; c < TT; ++c) sP[lane][c] = (_Float16)(sS[lane][c] * inv);
  }
  __syncthreads();

  // ctx = P V  (16 x 64)
  v8f acc[4] = {};
  for (int ks = 0; ks < TT / 32; ++ks) {
    const int k0 = ks * 32;
    v16h pa;
    #pragma unroll
    for (int e = 0; e < 8; ++e) {
      pa[e]     = sP[lrow][k0 + kb + e];
      pa[e + 8] = sP[lrow][k0 + kb + 8 + e];
    }
    #pragma unroll
    for (int nt = 0; nt < 4; ++nt) {
      const int n = nt * 16 + lrow;
      v16h bf;
      #pragma unroll
      for (int e = 0; e < 16; ++e)
        bf[e] = (_Float16)V[base + (k0 + kb + e) * DD + n];
      acc[nt] = __builtin_amdgcn_wmma_f32_16x16x32_f16(
          false, pa, false, bf, (short)0, acc[nt], false, false);
    }
  }
  #pragma unroll
  for (int nt = 0; nt < 4; ++nt)
    #pragma unroll
    for (int r = 0; r < 8; ++r)
      ctx[base + (qt * 16 + rbase + r) * DD + nt * 16 + lrow] = acc[nt][r];
}

// ======================= 5. (residual +) layernorm ==========================
template <bool ADD>
__global__ __launch_bounds__(256) void ln_kernel(
    const float* __restrict__ X, const float* __restrict__ Y,
    const float* __restrict__ g, const float* __restrict__ bta,
    float* __restrict__ out) {
  __shared__ float red[256];
  const int row = blockIdx.x, tid = threadIdx.x;
  const float* xr = X + row * DD;
  float a0 = xr[tid], a1 = xr[tid + 256];
  if (ADD) {
    const float* yr = Y + row * DD;
    a0 += yr[tid];
    a1 += yr[tid + 256];
  }
  red[tid] = a0 + a1;
  __syncthreads();
  for (int s = 128; s > 0; s >>= 1) {
    if (tid < s) red[tid] += red[tid + s];
    __syncthreads();
  }
  const float mean = red[0] * (1.0f / DD);
  __syncthreads();
  const float d0 = a0 - mean, d1 = a1 - mean;
  red[tid] = d0 * d0 + d1 * d1;
  __syncthreads();
  for (int s = 128; s > 0; s >>= 1) {
    if (tid < s) red[tid] += red[tid + s];
    __syncthreads();
  }
  const float inv = rsqrtf(red[0] * (1.0f / DD) + 1e-5f);
  out[row * DD + tid]       = d0 * inv * g[tid] + bta[tid];
  out[row * DD + tid + 256] = d1 * inv * g[tid + 256] + bta[tid + 256];
}

// ============ 6. final norm + ReLU MLP + trend recombination (t = T-1) ======
__global__ __launch_bounds__(256) void final_kernel(
    const float* __restrict__ enc, const float* __restrict__ norm_g,
    const float* __restrict__ norm_b, const float* __restrict__ mlp_w1,
    const float* __restrict__ mlp_b1, const float* __restrict__ mlp_w2,
    const float* __restrict__ mlp_b2, const float* __restrict__ w_dec,
    const float* __restrict__ trend_out, float* __restrict__ out) {
  __shared__ float row[DD];
  __shared__ float hid[DD];
  __shared__ float red[256];
  const int b = blockIdx.x, tid = threadIdx.x;
  const float* e = enc + (b * TT + (TT - 1)) * DD;
  float a0 = e[tid], a1 = e[tid + 256];
  red[tid] = a0 + a1;
  __syncthreads();
  for (int s = 128; s > 0; s >>= 1) {
    if (tid < s) red[tid] += red[tid + s];
    __syncthreads();
  }
  const float mean = red[0] * (1.0f / DD);
  __syncthreads();
  const float d0 = a0 - mean, d1 = a1 - mean;
  red[tid] = d0 * d0 + d1 * d1;
  __syncthreads();
  for (int s = 128; s > 0; s >>= 1) {
    if (tid < s) red[tid] += red[tid + s];
    __syncthreads();
  }
  const float inv = rsqrtf(red[0] * (1.0f / DD) + 1e-5f);
  row[tid]       = d0 * inv * norm_g[tid] + norm_b[tid];
  row[tid + 256] = d1 * inv * norm_g[tid + 256] + norm_b[tid + 256];
  __syncthreads();
  for (int j = tid; j < DD; j += 256) {
    float s = mlp_b1[j];
    for (int k = 0; k < DD; ++k) s += row[k] * mlp_w1[j * DD + k];
    hid[j] = fmaxf(s, 0.0f);
  }
  __syncthreads();
  red[tid] = hid[tid] * mlp_w2[tid] + hid[tid + 256] * mlp_w2[tid + 256];
  __syncthreads();
  for (int s = 128; s > 0; s >>= 1) {
    if (tid < s) red[tid] += red[tid + s];
    __syncthreads();
  }
  if (tid == 0)
    out[b] = red[0] + mlp_b2[0] + w_dec[0] * trend_out[b * FF + 0];
}

// ============================== launcher ====================================
extern "C" void kernel_launch(void* const* d_in, const int* in_sizes, int n_in,
                              void* d_out, int out_size, void* d_ws,
                              size_t ws_size, hipStream_t stream) {
  (void)in_sizes; (void)n_in; (void)out_size; (void)ws_size;
  const float* x       = (const float*)d_in[0];
  // d_in[1] = queue_ids — unused by the reference forward pass.
  const float* conv_w  = (const float*)d_in[2];
  const float* wq      = (const float*)d_in[3];
  const float* bq      = (const float*)d_in[4];
  const float* wk      = (const float*)d_in[5];
  const float* bk      = (const float*)d_in[6];
  const float* wv      = (const float*)d_in[7];
  const float* bv      = (const float*)d_in[8];
  const float* wo      = (const float*)d_in[9];
  const float* bo      = (const float*)d_in[10];
  const float* ln1_g   = (const float*)d_in[11];
  const float* ln1_b   = (const float*)d_in[12];
  const float* ln2_g   = (const float*)d_in[13];
  const float* ln2_b   = (const float*)d_in[14];
  const float* ffn_w1  = (const float*)d_in[15];
  const float* ffn_b1  = (const float*)d_in[16];
  const float* ffn_w2  = (const float*)d_in[17];
  const float* ffn_b2  = (const float*)d_in[18];
  const float* norm_g  = (const float*)d_in[19];
  const float* norm_b  = (const float*)d_in[20];
  const float* trend_w = (const float*)d_in[21];
  const float* trend_b = (const float*)d_in[22];
  const float* w_dec   = (const float*)d_in[23];
  const float* mlp_w1  = (const float*)d_in[24];
  const float* mlp_b1  = (const float*)d_in[25];
  const float* mlp_w2  = (const float*)d_in[26];
  const float* mlp_b2  = (const float*)d_in[27];
  float* out = (float*)d_out;

  // ---- workspace layout (floats) ----
  const size_t nTok = (size_t)BB * TT;            // 8192
  float* ws       = (float*)d_ws;
  float* seasonal = ws;                           // B*T*F
  float* trend_o  = seasonal + (size_t)BB * TT * FF;
  float* enc      = trend_o + 512;                // B*T*D
  float* bufQ     = enc   + nTok * DD;
  float* bufK     = bufQ  + nTok * DD;
  float* bufV     = bufK  + nTok * DD;
  float* bufH     = bufV  + nTok * DD;            // attention ctx
  float* bufF     = bufH  + nTok * DD;            // B*T*DFF
  // total ≈ 38.1M floats ≈ 153 MB

  // 1) decomposition + trend projection
  decomp_trend_kernel<<<dim3(BB), 32, 0, stream>>>(x, trend_w, trend_b,
                                                   seasonal, trend_o);
  // 2) conv embedding + positional encoding
  conv_pos_kernel<<<dim3(BB * TT), 256, 0, stream>>>(seasonal, conv_w, enc);

  const dim3 gD(nTok / BM, DD / BN);              // 128 x 4
  const dim3 gF(nTok / BM, DFFN / BN);            // 128 x 16

  for (int l = 0; l < NLAY; ++l) {
    const float* wq_l = wq + (size_t)l * DD * DD;
    const float* wk_l = wk + (size_t)l * DD * DD;
    const float* wv_l = wv + (size_t)l * DD * DD;
    const float* wo_l = wo + (size_t)l * DD * DD;
    const float* w1_l = ffn_w1 + (size_t)l * DFFN * DD;
    const float* w2_l = ffn_w2 + (size_t)l * DD * DFFN;

    // Q, K, V projections
    gemm_bias_act_kernel<0><<<gD, 256, 0, stream>>>(enc, DD, wq_l, DD,
                                                    bq + l * DD, bufQ, DD, DD);
    gemm_bias_act_kernel<0><<<gD, 256, 0, stream>>>(enc, DD, wk_l, DD,
                                                    bk + l * DD, bufK, DD, DD);
    gemm_bias_act_kernel<0><<<gD, 256, 0, stream>>>(enc, DD, wv_l, DD,
                                                    bv + l * DD, bufV, DD, DD);
    // attention -> ctx in bufH
    attention_kernel<<<dim3(TT / 16, HH, BB), 32, 0, stream>>>(bufQ, bufK,
                                                               bufV, bufH);
    // output projection: attn_out -> bufQ
    gemm_bias_act_kernel<0><<<gD, 256, 0, stream>>>(bufH, DD, wo_l, DD,
                                                    bo + l * DD, bufQ, DD, DD);
    // h1 = LN(enc + attn_out) -> bufK
    ln_kernel<true><<<dim3(nTok), 256, 0, stream>>>(enc, bufQ, ln1_g + l * DD,
                                                    ln1_b + l * DD, bufK);
    // y1 = gelu(h1 @ w1^T + b1) -> bufF
    gemm_bias_act_kernel<1><<<gF, 256, 0, stream>>>(bufK, DD, w1_l, DD,
                                                    ffn_b1 + l * DFFN, bufF,
                                                    DFFN, DD);
    // y2 = y1 @ w2^T + b2 -> bufV
    gemm_bias_act_kernel<0><<<gD, 256, 0, stream>>>(bufF, DFFN, w2_l, DFFN,
                                                    ffn_b2 + l * DD, bufV, DD,
                                                    DFFN);
    // enc = LN(h1 + y2)
    ln_kernel<true><<<dim3(nTok), 256, 0, stream>>>(bufK, bufV, ln2_g + l * DD,
                                                    ln2_b + l * DD, enc);
  }

  // final LN (only last token matters) + MLP head + trend
  final_kernel<<<dim3(BB), 256, 0, stream>>>(enc, norm_g, norm_b, mlp_w1,
                                             mlp_b1, mlp_w2, mlp_b2, w_dec,
                                             trend_o, out);
}
